// Reddit5kNet_21277267984767
// MI455X (gfx1250) — matrix operand order repaired and checked by reading, not compiled, not run on hardware
//
#include <hip/hip_runtime.h>

#define NN 200000
#define EE 2000000
#define GG 1000
#define DD 32
#define EPSV 1e-5f

typedef __attribute__((ext_vector_type(16))) _Float16 v16h;
typedef __attribute__((ext_vector_type(8)))  float    v8f;

// ---------------------------------------------------------------- utilities
__global__ void zero_kernel(float* __restrict__ p, long long n) {
    long long i = (long long)blockIdx.x * blockDim.x + threadIdx.x;
    if (i < n) p[i] = 0.0f;
}

// ------------------------------------------------------------- lin0 + tanh
__global__ void lin0_kernel(const float* __restrict__ x,
                            const float* __restrict__ w,
                            const float* __restrict__ b,
                            float* __restrict__ h) {
    long long i = (long long)blockIdx.x * blockDim.x + threadIdx.x;
    if (i >= (long long)NN * DD) return;
    int n = (int)(i >> 5);
    int d = (int)(i & 31);
    h[i] = tanhf(x[n] * w[d] + b[d]);
}

// --------------------------------------------------- edge scatter (atomic)
// one lane per (edge, dim): coalesced 128B gather of h[src], atomic scatter
__global__ void scatter_kernel(const long long* __restrict__ ei,
                               const float* __restrict__ ew,
                               const float* __restrict__ h,
                               float* __restrict__ agg) {
    long long idx = (long long)blockIdx.x * blockDim.x + threadIdx.x;
    long long e = idx >> 5;
    int d = (int)(idx & 31);
    if (e >= (long long)EE) return;
    long long s = ei[e];
    long long t = ei[(long long)EE + e];
    float v = ew[e] * h[s * DD + d];
    atomicAdd(&agg[t * DD + d], v);
}

// --------------------------------------- GraphConv node update via WMMA
// out[16x32 tile] = relu(agg @ rel_w^T + rel_b + h @ root_w^T)
// Two 16x16 N-tiles, K=32 exactly -> V_WMMA_F32_16X16X32_F16, f32 accumulate.
__global__ void conv_wmma_kernel(const float* __restrict__ agg,
                                 const float* __restrict__ hin,
                                 const float* __restrict__ rel_w,
                                 const float* __restrict__ rel_b,
                                 const float* __restrict__ root_w,
                                 float* __restrict__ hout) {
    const int lane = threadIdx.x & 31;
    const int wave = threadIdx.x >> 5;
    const int tile = blockIdx.x * (blockDim.x >> 5) + wave;
    if (tile >= NN / 16) return;            // uniform per wave (8 waves/block)
    const int row0 = tile * 16;
    const int m = lane & 15;                // row within tile for A, col for B/C
    const int hi = (lane >= 16) ? 1 : 0;

    // ---- A fragments: 16-bit A 16x32 layout
    // lanes 0-15: halves 0-7 -> K=0..7, halves 8-15 -> K=16..23
    // lanes 16-31: halves 0-7 -> K=8..15, halves 8-15 -> K=24..31
    v16h a_agg, a_h;
    const float* aggRow = agg + (long long)(row0 + m) * DD;
    const float* hRow   = hin + (long long)(row0 + m) * DD;
#pragma unroll
    for (int i = 0; i < 16; ++i) {
        int k = ((i < 8) ? i : (i + 8)) + (hi ? 8 : 0);
        a_agg[i] = (_Float16)aggRow[k];
        a_h[i]   = (_Float16)hRow[k];
    }

    // ---- B fragments: 16-bit B 32x16 layout, B[k][n] = W[n][k]
    // lanes 0-15: halves 0-7 -> K=0..7, 8-15 -> K=16..23; lanes 16-31: K=8..15, K=24..31
    const int kb = hi ? 16 : 0;
    v16h b_rel0, b_rel1, b_root0, b_root1;
#pragma unroll
    for (int i = 0; i < 16; ++i) {
        int k = ((i < 8) ? i : (i + 8)) + (hi ? 8 : 0);
        b_rel0[i]  = (_Float16)rel_w[m * DD + k];          // cols 0..15
        b_rel1[i]  = (_Float16)rel_w[(m + 16) * DD + k];   // cols 16..31
        b_root0[i] = (_Float16)root_w[m * DD + k];
        b_root1[i] = (_Float16)root_w[(m + 16) * DD + k];
    }

    v8f c0 = {}, c1 = {};
    c0 = __builtin_amdgcn_wmma_f32_16x16x32_f16(false, a_agg, false, b_rel0,
                                                (short)0, c0, false, false);
    c0 = __builtin_amdgcn_wmma_f32_16x16x32_f16(false, a_h,   false, b_root0,
                                                (short)0, c0, false, false);
    c1 = __builtin_amdgcn_wmma_f32_16x16x32_f16(false, a_agg, false, b_rel1,
                                                (short)0, c1, false, false);
    c1 = __builtin_amdgcn_wmma_f32_16x16x32_f16(false, a_h,   false, b_root1,
                                                (short)0, c1, false, false);

    // ---- C/D layout: VGPR r holds row M=r (lanes 0-15) / M=r+8 (lanes 16-31)
    const int mrow = hi ? 8 : 0;
    const float b0 = rel_b[m];
    const float b1 = rel_b[m + 16];
#pragma unroll
    for (int r = 0; r < 8; ++r) {
        long long row = row0 + r + mrow;
        float v0 = c0[r] + b0;  v0 = v0 > 0.0f ? v0 : 0.0f;
        float v1 = c1[r] + b1;  v1 = v1 > 0.0f ? v1 : 0.0f;
        hout[row * DD + m]      = v0;
        hout[row * DD + m + 16] = v1;
    }
}

// ---------------------------------------- per-column sum / sumsq (for BN)
__global__ void stats_kernel(const float* __restrict__ h,
                             float* __restrict__ stats) {
    __shared__ float ssum[256];
    __shared__ float ssq[256];
    const int d = blockIdx.x;   // 0..31
    float s = 0.0f, q = 0.0f;
    for (int n = threadIdx.x; n < NN; n += blockDim.x) {
        float v = h[(long long)n * DD + d];
        s += v; q += v * v;
    }
    ssum[threadIdx.x] = s; ssq[threadIdx.x] = q;
    __syncthreads();
    for (int off = 128; off > 0; off >>= 1) {
        if ((int)threadIdx.x < off) {
            ssum[threadIdx.x] += ssum[threadIdx.x + off];
            ssq[threadIdx.x]  += ssq[threadIdx.x + off];
        }
        __syncthreads();
    }
    if (threadIdx.x == 0) { stats[d] = ssum[0]; stats[DD + d] = ssq[0]; }
}

// ---------------------------------------------------- BN apply (in place)
__global__ void bn_kernel(float* __restrict__ h,
                          const float* __restrict__ stats,
                          const float* __restrict__ g,
                          const float* __restrict__ b) {
    long long i = (long long)blockIdx.x * blockDim.x + threadIdx.x;
    if (i >= (long long)NN * DD) return;
    int d = (int)(i & 31);
    float mean = stats[d] * (1.0f / (float)NN);
    float var  = stats[DD + d] * (1.0f / (float)NN) - mean * mean;
    float v = h[i];
    h[i] = (v - mean) * rsqrtf(var + EPSV) * g[d] + b[d];
}

// ------------------------------------------------- global mean pool (part1)
__global__ void pool_kernel(const float* __restrict__ h,
                            const long long* __restrict__ batch,
                            float* __restrict__ gsum,
                            float* __restrict__ gcnt) {
    long long idx = (long long)blockIdx.x * blockDim.x + threadIdx.x;
    long long n = idx >> 5;
    int d = (int)(idx & 31);
    if (n >= (long long)NN) return;
    long long g = batch[n];
    atomicAdd(&gsum[g * DD + d], h[n * DD + d]);
    if (d == 0) atomicAdd(&gcnt[g], 1.0f);
}

// -------------------------------------------------- MLP head, one blk/graph
__global__ void head_kernel(const float* __restrict__ gsum,
                            const float* __restrict__ gcnt,
                            const float* __restrict__ lin1_w,
                            const float* __restrict__ lin1_b,
                            const float* __restrict__ lin2_w,
                            const float* __restrict__ lin2_b,
                            float* __restrict__ out) {
    __shared__ float gx[DD];
    __shared__ float p[64];
    const int g = blockIdx.x;
    const int t = threadIdx.x;   // 64 threads
    if (t < DD) {
        float c = gcnt[g];
        c = c < 1.0f ? 1.0f : c;
        gx[t] = gsum[(long long)g * DD + t] / c;
    }
    __syncthreads();
    float acc = lin1_b[t];
#pragma unroll
    for (int d = 0; d < DD; ++d) acc += gx[d] * lin1_w[t * DD + d];
    p[t] = acc > 0.0f ? acc : 0.0f;
    __syncthreads();
    if (t < 5) {
        float acc2 = lin2_b[t];
#pragma unroll
        for (int j = 0; j < 64; ++j) acc2 += p[j] * lin2_w[t * 64 + j];
        out[(long long)g * 5 + t] = acc2;
    }
}

// ---------------------------------------------------------------- driver
extern "C" void kernel_launch(void* const* d_in, const int* in_sizes, int n_in,
                              void* d_out, int out_size, void* d_ws, size_t ws_size,
                              hipStream_t stream) {
    (void)in_sizes; (void)n_in; (void)out_size; (void)ws_size;
    const float*      x        = (const float*)d_in[0];
    const long long*  ei       = (const long long*)d_in[1];   // int64 per reference
    const float*      ew       = (const float*)d_in[2];
    const long long*  batch    = (const long long*)d_in[3];
    const float*      lin0_w   = (const float*)d_in[4];
    const float*      lin0_b   = (const float*)d_in[5];
    const float*      rel_w0   = (const float*)d_in[6];
    const float*      rel_b0   = (const float*)d_in[7];
    const float*      root_w0  = (const float*)d_in[8];
    const float*      rel_w1   = (const float*)d_in[9];
    const float*      rel_b1   = (const float*)d_in[10];
    const float*      root_w1  = (const float*)d_in[11];
    const float*      bn_g0    = (const float*)d_in[12];
    const float*      bn_b0    = (const float*)d_in[13];
    const float*      bn_g1    = (const float*)d_in[14];
    const float*      bn_b1    = (const float*)d_in[15];
    const float*      lin1_w   = (const float*)d_in[16];
    const float*      lin1_b   = (const float*)d_in[17];
    const float*      lin2_w   = (const float*)d_in[18];
    const float*      lin2_b   = (const float*)d_in[19];
    float* out = (float*)d_out;

    // workspace layout (floats)
    float* ws    = (float*)d_ws;
    const long long ND = (long long)NN * DD;
    float* h_a   = ws;                 // N*D
    float* agg   = h_a + ND;           // N*D
    float* h_b   = agg + ND;           // N*D
    float* stats = h_b + ND;           // 64
    float* gsum  = stats + 64;         // G*D
    float* gcnt  = gsum + (long long)GG * DD;  // G

    const int T = 256;
    const int ndBlocks   = (int)((ND + T - 1) / T);
    const int edgeBlocks = (int)(((long long)EE * 32 + T - 1) / T);
    const int convBlocks = (NN / 16 + 7) / 8;          // 8 waves per block
    const int poolBlocks = (int)(((long long)NN * 32 + T - 1) / T);
    const int gBlocks    = (int)(((long long)GG * DD + GG + T - 1) / T);

    // h0 = tanh(x @ lin0_w.T + lin0_b)
    lin0_kernel<<<ndBlocks, T, 0, stream>>>(x, lin0_w, lin0_b, h_a);

    // ---- layer 0
    zero_kernel<<<ndBlocks, T, 0, stream>>>(agg, ND);
    scatter_kernel<<<edgeBlocks, T, 0, stream>>>(ei, ew, h_a, agg);
    conv_wmma_kernel<<<convBlocks, T, 0, stream>>>(agg, h_a, rel_w0, rel_b0,
                                                   root_w0, h_b);
    stats_kernel<<<DD, T, 0, stream>>>(h_b, stats);
    bn_kernel<<<ndBlocks, T, 0, stream>>>(h_b, stats, bn_g0, bn_b0);

    // ---- layer 1
    zero_kernel<<<ndBlocks, T, 0, stream>>>(agg, ND);
    scatter_kernel<<<edgeBlocks, T, 0, stream>>>(ei, ew, h_b, agg);
    conv_wmma_kernel<<<convBlocks, T, 0, stream>>>(agg, h_b, rel_w1, rel_b1,
                                                   root_w1, h_a);
    stats_kernel<<<DD, T, 0, stream>>>(h_a, stats);
    bn_kernel<<<ndBlocks, T, 0, stream>>>(h_a, stats, bn_g1, bn_b1);

    // ---- global mean pool + head
    zero_kernel<<<gBlocks, T, 0, stream>>>(gsum, (long long)GG * DD + GG);
    pool_kernel<<<poolBlocks, T, 0, stream>>>(h_a, batch, gsum, gcnt);
    head_kernel<<<GG, 64, 0, stream>>>(gsum, gcnt, lin1_w, lin1_b,
                                       lin2_w, lin2_b, out);
}
